// CausalSelfAttention_64390149702272
// MI455X (gfx1250) — matrix operand order repaired and checked by reading, not compiled, not run on hardware
//
#include <hip/hip_runtime.h>

// ---------------------------------------------------------------------------
// Types / helpers
// ---------------------------------------------------------------------------
typedef unsigned short u16;
typedef u16    u16x8   __attribute__((ext_vector_type(8)));
typedef u16    u16x16  __attribute__((ext_vector_type(16)));
typedef __bf16 bf16x16 __attribute__((ext_vector_type(16)));
typedef float  f32x8   __attribute__((ext_vector_type(8)));

#define NEG_INF (-3.0e38f)

static __device__ __forceinline__ u16 f2bf(float f) {
  unsigned u = __float_as_uint(f);
  unsigned r = (u + 0x7FFFu + ((u >> 16) & 1u)) >> 16;  // round-to-nearest-even
  return (u16)r;
}

static __device__ __forceinline__ bf16x16 mk_frag(u16x8 lo, u16x8 hi) {
  u16x16 r;
#pragma unroll
  for (int i = 0; i < 8; ++i) { r[i] = lo[i]; r[i + 8] = hi[i]; }
  return __builtin_bit_cast(bf16x16, r);
}

static __device__ __forceinline__ f32x8 zero8() {
  f32x8 z;
#pragma unroll
  for (int i = 0; i < 8; ++i) z[i] = 0.f;
  return z;
}

static __device__ __forceinline__ f32x8 wmma_bf16(bf16x16 a, bf16x16 b, f32x8 c) {
  // D = A(16x32 bf16) x B(32x16 bf16) + C(16x16 f32)
  return __builtin_amdgcn_wmma_f32_16x16x32_bf16(false, a, false, b, (short)0, c,
                                                 false, false);
}

// CDNA5 async copy global->LDS (16 bytes/lane), tracked by ASYNCcnt.
// LDS address = low 32 bits of the generic pointer (aperture layout, ISA 10.2).
static __device__ __forceinline__ void async_ld_b128(unsigned lds_addr,
                                                     const void* gptr) {
  asm volatile("global_load_async_to_lds_b128 %0, %1, off"
               :
               : "v"(lds_addr), "v"((unsigned long long)(uintptr_t)gptr)
               : "memory");
}
static __device__ __forceinline__ void async_wait0() {
  asm volatile("s_wait_asynccnt 0x0" ::: "memory");
}
static __device__ __forceinline__ unsigned lds_off(const void* p) {
  return (unsigned)(uintptr_t)p;  // low 32 bits = LDS byte offset
}

// ---------------------------------------------------------------------------
// fp32 -> bf16 conversion
// ---------------------------------------------------------------------------
__global__ void f32_to_bf16_k(const float* __restrict__ in, u16* __restrict__ out,
                              int n) {
  int i = blockIdx.x * blockDim.x + threadIdx.x;
  if (i < n) out[i] = f2bf(in[i]);
}

// ---------------------------------------------------------------------------
// Tiled bf16 WMMA GEMM: C[M,N] = A[M,K] * B[K,N]
//   MODE 0: N==2304, scatter into Q/K/V [B=2,H=12,S=4096,hd=64] as bf16
//   MODE 1: write fp32 to outF[M,N]
// Block: 256 threads (8 waves), tile 128x128, BK=32.
// A tile staged with global_load_async_to_lds_b128 (row-major, no transpose).
// ---------------------------------------------------------------------------
template <int MODE>
__global__ __launch_bounds__(256) void gemm_bf16_k(
    const u16* __restrict__ A, const u16* __restrict__ Bmat, int N, int K,
    float* __restrict__ outF, u16* __restrict__ oq, u16* __restrict__ ok,
    u16* __restrict__ ov) {
  __shared__ __attribute__((aligned(16))) u16 As[128 * 32];  // [m][k]
  __shared__ __attribute__((aligned(16))) u16 Bt[128 * 32];  // [n][k] (transposed)

  const int tid  = threadIdx.x;
  const int lane = tid & 31;
  const int wv   = tid >> 5;
  const int wm   = wv & 3;   // wave M index (0..3)
  const int wn   = wv >> 2;  // wave N index (0..1)
  const int half = lane >> 4;
  const int lm   = lane & 15;
  const int row0 = blockIdx.y * 128;
  const int col0 = blockIdx.x * 128;

  f32x8 acc[2][4];
#pragma unroll
  for (int i = 0; i < 2; ++i)
#pragma unroll
    for (int j = 0; j < 4; ++j) acc[i][j] = zero8();

  const int ar = tid >> 1, ah = tid & 1;  // A stage: row 0..127, 16-elem half
  const int bk = tid >> 3, bc = tid & 7;  // B stage: k 0..31, col chunk 0..7

  for (int kb = 0; kb < K; kb += 32) {
    __syncthreads();
    {  // A tile -> LDS row-major [128][32], async (2 x b128 per thread)
      const u16* g = A + (size_t)(row0 + ar) * K + kb + ah * 16;
      const unsigned l = lds_off(As + ar * 32 + ah * 16);
      async_ld_b128(l, g);
      async_ld_b128(l + 16, g + 8);
    }
    {  // B tile -> LDS transposed [128 n][32 k]
      const u16x8* g =
          (const u16x8*)(Bmat + (size_t)(kb + bk) * N + col0 + bc * 16);
      u16x8 t0 = g[0], t1 = g[1];
#pragma unroll
      for (int i = 0; i < 8; ++i) {
        Bt[(bc * 16 + i) * 32 + bk]     = t0[i];
        Bt[(bc * 16 + 8 + i) * 32 + bk] = t1[i];
      }
    }
    async_wait0();
    __syncthreads();

    // A frag: lane(0-15)=row M, elem e: K = half*8+e (e<8) / 16+half*8+(e-8)
    bf16x16 af[2];
#pragma unroll
    for (int i = 0; i < 2; ++i) {
      const u16* base = As + (wm * 32 + i * 16 + lm) * 32;
      af[i] = mk_frag(*(const u16x8*)(base + half * 8),
                      *(const u16x8*)(base + 16 + half * 8));
    }
    // B frag: lane(0-15)=col N, elem e: K = half*8+e (e<8) / 16+half*8+(e-8)
    bf16x16 bfr[4];
#pragma unroll
    for (int j = 0; j < 4; ++j) {
      const u16* base = Bt + (wn * 64 + j * 16 + lm) * 32;
      bfr[j] = mk_frag(*(const u16x8*)(base + half * 8),
                       *(const u16x8*)(base + 16 + half * 8));
    }
#pragma unroll
    for (int i = 0; i < 2; ++i)
#pragma unroll
      for (int j = 0; j < 4; ++j) acc[i][j] = wmma_bf16(af[i], bfr[j], acc[i][j]);
  }

  // Epilogue. D layout: VGPR e, lane l -> m = e + 8*(l>>4), n = l&15.
#pragma unroll
  for (int i = 0; i < 2; ++i) {
#pragma unroll
    for (int j = 0; j < 4; ++j) {
#pragma unroll
      for (int e = 0; e < 8; ++e) {
        const int m = row0 + wm * 32 + i * 16 + e + 8 * half;
        const int n = col0 + wn * 64 + j * 16 + lm;
        const float val = acc[i][j][e];
        if constexpr (MODE == 0) {
          const int which = n / 768;
          const int d     = n % 768;
          const int h     = d >> 6;
          const int dd    = d & 63;
          u16* dst = (which == 0) ? oq : (which == 1) ? ok : ov;
          const int bb = m >> 12, ss = m & 4095;
          dst[((((size_t)bb * 12 + h) * 4096) + ss) * 64 + dd] = f2bf(val);
        } else {
          outF[(size_t)m * N + n] = val;
        }
      }
    }
  }
}

// ---------------------------------------------------------------------------
// Flash attention, causal. Q/K/V bf16 [2,12,4096,64]. Ctx bf16 [8192,768].
// Block: 128 threads (4 waves); block covers 64 query rows, wave covers 16.
// Key blocks of 64 (16 WMMAs per iter); loop count uniform per block.
// K tile staged via async-to-LDS; V staged transposed via VGPR bounce.
// ---------------------------------------------------------------------------
__global__ __launch_bounds__(128) void attn_k(const u16* __restrict__ Qg,
                                              const u16* __restrict__ Kg,
                                              const u16* __restrict__ Vg,
                                              u16* __restrict__ Ctx) {
  __shared__ __attribute__((aligned(16))) u16 Ks[64 * 64];      // [key][hd]
  __shared__ __attribute__((aligned(16))) u16 Vt[64 * 64];      // [hd][key]
  __shared__ __attribute__((aligned(16))) u16 Ps[4 * 16 * 64];  // per-wave P

  const int tid  = threadIdx.x;
  const int lane = tid & 31;
  const int wv   = tid >> 5;
  const int half = lane >> 4;
  const int lm   = lane & 15;

  const int q0 = blockIdx.x * 64;
  const int qt = q0 + wv * 16;
  const size_t bh = (((size_t)blockIdx.z * 12 + blockIdx.y) * 4096) * 64;

  // Q A-frags (rows qt..qt+15, hd split 0..31 / 32..63), direct from global
  bf16x16 aQ[2];
#pragma unroll
  for (int kc = 0; kc < 2; ++kc) {
    const u16* base = Qg + bh + (size_t)(qt + lm) * 64 + kc * 32;
    aQ[kc] = mk_frag(*(const u16x8*)(base + half * 8),
                     *(const u16x8*)(base + 16 + half * 8));
  }

  f32x8 acc[4];
#pragma unroll
  for (int j = 0; j < 4; ++j) acc[j] = zero8();
  float mrow[8], lrow[8];
#pragma unroll
  for (int e = 0; e < 8; ++e) { mrow[e] = NEG_INF; lrow[e] = 0.f; }

  const int nblk = q0 / 64 + 1;  // keys 0 .. q0+63
  const int ld_r = tid >> 1;     // key row 0..63
  const int ld_c = tid & 1;      // 32-elem half of hd

  for (int jb = 0; jb < nblk; ++jb) {
    const int key0 = jb * 64;
    __syncthreads();
    {  // stage K row-major via async-to-LDS (4 x b128 per thread)
      const u16* g = Kg + bh + (size_t)(key0 + ld_r) * 64 + ld_c * 32;
      const unsigned l = lds_off(Ks + ld_r * 64 + ld_c * 32);
#pragma unroll
      for (int ch = 0; ch < 4; ++ch) async_ld_b128(l + ch * 16, g + ch * 8);
    }
    {  // stage V transposed [hd][key]
      const u16x8* g =
          (const u16x8*)(Vg + bh + (size_t)(key0 + ld_r) * 64 + ld_c * 32);
#pragma unroll
      for (int ch = 0; ch < 4; ++ch) {
        u16x8 v = g[ch];
#pragma unroll
        for (int i = 0; i < 8; ++i)
          Vt[(ld_c * 32 + ch * 8 + i) * 64 + ld_r] = v[i];
      }
    }
    async_wait0();
    __syncthreads();

    // scores: S = Q @ K^T  (four 16x16 tiles over 64 keys, K-dim = hd = 64)
    f32x8 s[4];
#pragma unroll
    for (int t = 0; t < 4; ++t) s[t] = zero8();
#pragma unroll
    for (int t = 0; t < 4; ++t) {
#pragma unroll
      for (int kc = 0; kc < 2; ++kc) {
        const u16* b = Ks + (t * 16 + lm) * 64 + kc * 32;
        s[t] = wmma_bf16(aQ[kc],
                         mk_frag(*(const u16x8*)(b + half * 8), *(const u16x8*)(b + 16 + half * 8)), s[t]);
      }
    }

    // scale + causal mask + online softmax (row stats per 16-lane half)
#pragma unroll
    for (int e = 0; e < 8; ++e) {
      const int qrow = qt + e + 8 * half;
      float sv[4];
#pragma unroll
      for (int t = 0; t < 4; ++t) {
        float v = s[t][e] * 0.125f;  // 1/sqrt(64)
        sv[t] = ((key0 + t * 16 + lm) <= qrow) ? v : NEG_INF;
      }
      float tmax = fmaxf(fmaxf(sv[0], sv[1]), fmaxf(sv[2], sv[3]));
#pragma unroll
      for (int off = 1; off < 16; off <<= 1)
        tmax = fmaxf(tmax, __shfl_xor(tmax, off, 32));
      const float mnew  = fmaxf(mrow[e], tmax);
      const float alpha = __expf(mrow[e] - mnew);
      mrow[e] = mnew;
      float ls = 0.f;
#pragma unroll
      for (int t = 0; t < 4; ++t) {
        const float p = __expf(sv[t] - mnew);
        s[t][e] = p;
        ls += p;
      }
#pragma unroll
      for (int off = 1; off < 16; off <<= 1) ls += __shfl_xor(ls, off, 32);
      lrow[e] = lrow[e] * alpha + ls;
#pragma unroll
      for (int j = 0; j < 4; ++j) acc[j][e] *= alpha;
    }

    // re-layout P (D layout) -> A-frag layout via per-wave LDS bounce
    u16* ps = Ps + wv * (16 * 64);
#pragma unroll
    for (int e = 0; e < 8; ++e) {
#pragma unroll
      for (int t = 0; t < 4; ++t)
        ps[(e + 8 * half) * 64 + t * 16 + lm] = f2bf(s[t][e]);
    }
    __syncthreads();

    // ctx += P(16x64) @ V(64x64): two 32-key chunks
#pragma unroll
    for (int kc = 0; kc < 2; ++kc) {
      const u16* pb = ps + lm * 64 + kc * 32;
      bf16x16 aP = mk_frag(*(const u16x8*)(pb + half * 8),
                           *(const u16x8*)(pb + 16 + half * 8));
#pragma unroll
      for (int j = 0; j < 4; ++j) {
        const u16* vb = Vt + (j * 16 + lm) * 64 + kc * 32;
        acc[j] = wmma_bf16(
            aP, mk_frag(*(const u16x8*)(vb + half * 8), *(const u16x8*)(vb + 16 + half * 8)), acc[j]);
      }
    }
  }

  // normalize + store ctx as bf16 [8192, 768]
#pragma unroll
  for (int j = 0; j < 4; ++j) {
#pragma unroll
    for (int e = 0; e < 8; ++e) {
      const int qrow = qt + e + 8 * half;
      const int n    = j * 16 + lm;
      const float l  = lrow[e];
      const float v  = (l > 0.f) ? acc[j][e] / l : 0.f;
      Ctx[((size_t)blockIdx.z * 4096 + qrow) * 768 + blockIdx.y * 64 + n] =
          f2bf(v);
    }
  }
}

// ---------------------------------------------------------------------------
// Launch
// ---------------------------------------------------------------------------
extern "C" void kernel_launch(void* const* d_in, const int* in_sizes, int n_in,
                              void* d_out, int out_size, void* d_ws,
                              size_t ws_size, hipStream_t stream) {
  (void)in_sizes; (void)n_in; (void)out_size; (void)ws_size;
  const float* x     = (const float*)d_in[0];  // [2,4096,768]
  const float* w_qkv = (const float*)d_in[1];  // [768,2304]
  const float* w_out = (const float*)d_in[2];  // [768,768]
  float* out = (float*)d_out;                  // [2,4096,768]

  constexpr int NX    = 2 * 4096 * 768;  // 6291456
  constexpr int NWQKV = 768 * 2304;      // 1769472
  constexpr int NWOUT = 768 * 768;       // 589824
  constexpr int NQKV  = 2 * 12 * 4096 * 64;  // 6291456

  u16* Xbf    = (u16*)d_ws;
  u16* Wqkvbf = Xbf + NX;
  u16* Woutbf = Wqkvbf + NWQKV;
  u16* Qbf    = Woutbf + NWOUT;
  u16* Kbf    = Qbf + NQKV;
  u16* Vbf    = Kbf + NQKV;
  u16* Ctxbf  = Xbf;  // reuse: X is dead after the QKV GEMM

  f32_to_bf16_k<<<(NX + 255) / 256, 256, 0, stream>>>(x, Xbf, NX);
  f32_to_bf16_k<<<(NWQKV + 255) / 256, 256, 0, stream>>>(w_qkv, Wqkvbf, NWQKV);
  f32_to_bf16_k<<<(NWOUT + 255) / 256, 256, 0, stream>>>(w_out, Woutbf, NWOUT);

  // QKV projection: [8192,768] x [768,2304] -> Q/K/V [2,12,4096,64]
  dim3 gq(2304 / 128, 8192 / 128);
  gemm_bf16_k<0><<<gq, 256, 0, stream>>>(Xbf, Wqkvbf, 2304, 768, nullptr, Qbf,
                                         Kbf, Vbf);

  // causal flash attention -> ctx bf16 [8192,768]
  dim3 ga(4096 / 64, 12, 2);
  attn_k<<<ga, 128, 0, stream>>>(Qbf, Kbf, Vbf, Ctxbf);

  // output projection: [8192,768] x [768,768] -> fp32 out
  dim3 go(768 / 128, 8192 / 128);
  gemm_bf16_k<1><<<go, 256, 0, stream>>>(Ctxbf, Woutbf, 768, 768, out, nullptr,
                                         nullptr, nullptr);
}
